// GraphSAGE3_60352880443980
// MI455X (gfx1250) — matrix operand order repaired and checked by reading, not compiled, not run on hardware
//
#include <hip/hip_runtime.h>
#include <hip/hip_bf16.h>
#include <math.h>

// GraphSAGE-3 for MI455X (gfx1250, wave32, WMMA).
// Per layer: scatter-mean (global_atomic_add_f32, L2-resident) ->
// scale+cvt pass (agg*1/deg -> f16) ->
// fused dual-GEMM (v_wmma_f32_16x16x32_f16, A tiles staged in LDS)
// + bias + L2-normalize + ReLU. Final layer (N=2) is per-node VALU + log_softmax.

typedef __attribute__((ext_vector_type(16))) _Float16 v16h;
typedef __attribute__((ext_vector_type(8)))  _Float16 v8h;
typedef __attribute__((ext_vector_type(8)))  float    v8f;
typedef __attribute__((ext_vector_type(4)))  float    v4f;

#define NN  100000
#define NE  640000
#define D0  128
#define H1  256
#define H2  128

static __device__ __forceinline__ void atomic_add_f32(float* p, float v) {
  __hip_atomic_fetch_add(p, v, __ATOMIC_RELAXED, __HIP_MEMORY_SCOPE_AGENT);
}

__global__ void k_zero_f32(float* __restrict__ p, long long n) {
  long long i = (long long)blockIdx.x * blockDim.x + threadIdx.x;
  if (i < n) p[i] = 0.0f;
}

__global__ void k_f32_to_f16(const float* __restrict__ in, _Float16* __restrict__ out, long long n) {
  long long i = (long long)blockIdx.x * blockDim.x + threadIdx.x;
  if (i < n) out[i] = (_Float16)in[i];
}

__global__ void k_degree(const int* __restrict__ ei, float* __restrict__ deg, int E) {
  int e = blockIdx.x * blockDim.x + threadIdx.x;
  if (e < E) atomic_add_f32(&deg[ei[E + e]], 1.0f);
}

__global__ void k_deg_inv(float* __restrict__ deg, int n) {
  int i = blockIdx.x * blockDim.x + threadIdx.x;
  if (i < n) deg[i] = 1.0f / fmaxf(deg[i], 1.0f);
}

// Scatter-add of f16 features into f32 accumulator: one thread = 8 features of one edge.
__global__ void k_scatter(const _Float16* __restrict__ feat, const int* __restrict__ ei,
                          float* __restrict__ agg, int E, int dim) {
  const int chunks = dim >> 3;
  long long idx = (long long)blockIdx.x * blockDim.x + threadIdx.x;
  if (idx >= (long long)E * chunks) return;
  int e = (int)(idx / chunks);
  int c = (int)(idx - (long long)e * chunks);
  int s = ei[e];
  int d = ei[E + e];
  v8h v = *(const v8h*)(feat + (size_t)s * dim + (size_t)c * 8);
  float* ap = agg + (size_t)d * dim + (size_t)c * 8;
#pragma unroll
  for (int i = 0; i < 8; ++i) atomic_add_f32(ap + i, (float)v[i]);
}

// agg(f32) * (1/deg) -> f16, one pass. Removes all conversion VALU from the GEMM loop.
__global__ void k_scale_cvt(const float* __restrict__ agg, const float* __restrict__ dinv,
                            _Float16* __restrict__ out, long long rows, int dim) {
  const int chunks = dim >> 3;
  long long idx = (long long)blockIdx.x * blockDim.x + threadIdx.x;
  if (idx >= rows * chunks) return;
  long long row = idx / chunks;
  int c = (int)(idx - row * chunks) * 8;
  const float* p = agg + row * dim + c;
  float s = dinv[row];
  v4f a = *(const v4f*)p;
  v4f b = *(const v4f*)(p + 4);
  v8h o;
#pragma unroll
  for (int i = 0; i < 4; ++i) {
    o[i]     = (_Float16)(a[i] * s);
    o[4 + i] = (_Float16)(b[i] * s);
  }
  *(v8h*)(out + row * dim + c) = o;
}

// Fused SAGE layer: h = aggh@Wl.T + b + x@Wr.T ; out = relu(h / max(||h||,eps)) as f16.
// One block = 16 output rows x full N columns; wave w owns columns [w*16, w*16+16).
// A tiles (x and aggh, 16 x K f16 each) are staged once in LDS (stride K+8 halfs ->
// 528B row pitch -> conflict-free ds_load_b128) and reused by all waves.
// M = 100000 = 6250*16, K in {128,256}, N in {128,256}: no bounds checks, EXEC all-ones.
__global__ void __launch_bounds__(512)
k_sage_gemm_norm(const _Float16* __restrict__ feat, const _Float16* __restrict__ aggh,
                 const _Float16* __restrict__ Wl, const _Float16* __restrict__ Wr,
                 const float* __restrict__ bias, _Float16* __restrict__ out,
                 int K, int N) {
  __shared__ _Float16 lds_x[16 * 264];   // K<=256, stride K+8
  __shared__ _Float16 lds_g[16 * 264];
  __shared__ float ldsb[16 * 260];
  __shared__ float rscale[16];
  const int NS = 260;
  const int KS = K + 8;

  const int tid  = threadIdx.x;
  const int lane = tid & 31;
  const int wave = tid >> 5;
  const int mtile = blockIdx.x * 16;
  const int l15 = lane & 15;
  const int hi  = (lane >> 4) & 1;     // lane group 0..15 vs 16..31
  const int ka  = hi * 16;             // A K-offset base (contiguous k = 16*hi + s, matches B)
  const int kb0 = hi * 16;             // B K-offset base (contiguous k = 16*hi + s, same as A)
  const int ncol = wave * 16 + l15;    // output column

  // Cooperative stage of both A tiles into LDS (each thread moves 8-half chunks)
  const int kc8 = K >> 3;
  for (int idx = tid; idx < 16 * kc8; idx += blockDim.x) {
    int r = idx / kc8;
    int c = (idx - r * kc8) * 8;
    *(v8h*)(lds_x + r * KS + c) = *(const v8h*)(feat + (size_t)(mtile + r) * K + c);
    *(v8h*)(lds_g + r * KS + c) = *(const v8h*)(aggh + (size_t)(mtile + r) * K + c);
  }
  __syncthreads();

  const _Float16* pbl = Wl + (size_t)ncol * K + kb0;   // B column = weight row (K-major)
  const _Float16* pbr = Wr + (size_t)ncol * K + kb0;
  const _Float16* lx = lds_x + l15 * KS + ka;
  const _Float16* lg = lds_g + l15 * KS + ka;

  v8f acc = {};
#pragma unroll 2
  for (int kb = 0; kb < K; kb += 32) {
    v8h f0 = *(const v8h*)(lx + kb);
    v8h f1 = *(const v8h*)(lx + kb + 8);
    v8h g0 = *(const v8h*)(lg + kb);
    v8h g1 = *(const v8h*)(lg + kb + 8);
    v16h ax = __builtin_shufflevector(f0, f1, 0, 1, 2, 3, 4, 5, 6, 7,
                                              8, 9, 10, 11, 12, 13, 14, 15);
    v16h ag = __builtin_shufflevector(g0, g1, 0, 1, 2, 3, 4, 5, 6, 7,
                                              8, 9, 10, 11, 12, 13, 14, 15);
    v16h wl = *(const v16h*)(pbl + kb);
    v16h wr = *(const v16h*)(pbr + kb);

    acc = __builtin_amdgcn_wmma_f32_16x16x32_f16(false, ag, false, wl,
                                                 (short)0, acc, false, false);
    acc = __builtin_amdgcn_wmma_f32_16x16x32_f16(false, ax, false, wr,
                                                 (short)0, acc, false, false);
  }

  // C/D layout: VGPR v -> row (hi? v+8 : v), column l15. Bias is per-column.
  const float bn = bias[ncol];
  const int mbase = hi * 8;
#pragma unroll
  for (int v = 0; v < 8; ++v)
    ldsb[(mbase + v) * NS + ncol] = acc[v] + bn;
  __syncthreads();

  // Per-row L2 norm over the full N columns
  if (tid < 16) {
    float s = 0.0f;
    for (int n = 0; n < N; ++n) {
      float t = ldsb[tid * NS + n];
      s += t * t;
    }
    rscale[tid] = 1.0f / fmaxf(sqrtf(s), 1e-12f);
  }
  __syncthreads();

  // Normalize + ReLU + store as f16 for the next layer
  for (int idx = tid; idx < 16 * N; idx += blockDim.x) {
    int m = idx / N;
    int n = idx - m * N;
    float v = ldsb[m * NS + n] * rscale[m];
    out[(size_t)(mtile + m) * N + n] = (_Float16)fmaxf(v, 0.0f);
  }
}

// Final layer: N=2 outputs per node -> per-thread dot products (aggh pre-scaled f16),
// then L2-normalize over the 2 logits and log_softmax.
__global__ void k_layer3(const _Float16* __restrict__ feat, const _Float16* __restrict__ aggh,
                         const float* __restrict__ W3l, const float* __restrict__ W3r,
                         const float* __restrict__ b3, float* __restrict__ out,
                         int n, int K) {
  int i = blockIdx.x * blockDim.x + threadIdx.x;
  if (i >= n) return;
  const _Float16* f = feat + (size_t)i * K;
  const _Float16* g = aggh + (size_t)i * K;
  float h0 = b3[0], h1 = b3[1];
#pragma unroll 4
  for (int k = 0; k < K; ++k) {
    float a = (float)g[k];
    float x = (float)f[k];
    h0 += a * W3l[k]     + x * W3r[k];
    h1 += a * W3l[K + k] + x * W3r[K + k];
  }
  float nrm = sqrtf(h0 * h0 + h1 * h1);
  float sc  = 1.0f / fmaxf(nrm, 1e-12f);
  h0 *= sc; h1 *= sc;
  float m   = fmaxf(h0, h1);
  float lse = m + logf(expf(h0 - m) + expf(h1 - m));
  out[(size_t)i * 2]     = h0 - lse;
  out[(size_t)i * 2 + 1] = h1 - lse;
}

extern "C" void kernel_launch(void* const* d_in, const int* in_sizes, int n_in,
                              void* d_out, int out_size, void* d_ws, size_t ws_size,
                              hipStream_t stream) {
  (void)in_sizes; (void)n_in; (void)out_size; (void)ws_size;
  const float* x   = (const float*)d_in[0];
  const int*   ei  = (const int*)d_in[1];
  const float* W1l = (const float*)d_in[2];
  const float* W1r = (const float*)d_in[3];
  const float* b1  = (const float*)d_in[4];
  const float* W2l = (const float*)d_in[5];
  const float* W2r = (const float*)d_in[6];
  const float* b2  = (const float*)d_in[7];
  const float* W3l = (const float*)d_in[8];
  const float* W3r = (const float*)d_in[9];
  const float* b3  = (const float*)d_in[10];
  float* out = (float*)d_out;

  // Workspace carve (~257 MB)
  char* w = (char*)d_ws;
  size_t off = 0;
  auto carve = [&](size_t bytes) -> char* {
    char* p = w + off;
    off = (off + bytes + 255) & ~(size_t)255;
    return p;
  };
  float*    deg   = (float*)carve(sizeof(float) * NN);
  _Float16* w1l16 = (_Float16*)carve((size_t)2 * H1 * D0);
  _Float16* w1r16 = (_Float16*)carve((size_t)2 * H1 * D0);
  _Float16* w2l16 = (_Float16*)carve((size_t)2 * H2 * H1);
  _Float16* w2r16 = (_Float16*)carve((size_t)2 * H2 * H1);
  _Float16* bufA  = (_Float16*)carve((size_t)2 * NN * H1);
  _Float16* bufB  = (_Float16*)carve((size_t)2 * NN * H1);
  float*    agg   = (float*)carve((size_t)4 * NN * H1);
  _Float16* aggh  = (_Float16*)carve((size_t)2 * NN * H1);

  const int T = 256;
  // Degrees (shared across all three layers)
  k_zero_f32<<<(NN + T - 1) / T, T, 0, stream>>>(deg, NN);
  k_degree<<<(NE + T - 1) / T, T, 0, stream>>>(ei, deg, NE);
  k_deg_inv<<<(NN + T - 1) / T, T, 0, stream>>>(deg, NN);

  // Weights -> f16 (once)
  k_f32_to_f16<<<(H1 * D0 + T - 1) / T, T, 0, stream>>>(W1l, w1l16, (long long)H1 * D0);
  k_f32_to_f16<<<(H1 * D0 + T - 1) / T, T, 0, stream>>>(W1r, w1r16, (long long)H1 * D0);
  k_f32_to_f16<<<(H2 * H1 + T - 1) / T, T, 0, stream>>>(W2l, w2l16, (long long)H2 * H1);
  k_f32_to_f16<<<(H2 * H1 + T - 1) / T, T, 0, stream>>>(W2r, w2r16, (long long)H2 * H1);

  // x -> f16 into bufA (layer-1 input, stride 128)
  {
    long long n = (long long)NN * D0;
    k_f32_to_f16<<<(unsigned)((n + T - 1) / T), T, 0, stream>>>(x, bufA, n);
  }

  // ---- Layer 1: in bufA (128) -> out bufB (256)
  k_zero_f32<<<(unsigned)(((long long)NN * D0 + T - 1) / T), T, 0, stream>>>(agg, (long long)NN * D0);
  k_scatter<<<(unsigned)(((long long)NE * (D0 / 8) + T - 1) / T), T, 0, stream>>>(bufA, ei, agg, NE, D0);
  k_scale_cvt<<<(unsigned)(((long long)NN * (D0 / 8) + T - 1) / T), T, 0, stream>>>(agg, deg, aggh, NN, D0);
  k_sage_gemm_norm<<<NN / 16, (H1 / 16) * 32, 0, stream>>>(bufA, aggh, w1l16, w1r16, b1, bufB, D0, H1);

  // ---- Layer 2: in bufB (256) -> out bufA (128)
  k_zero_f32<<<(unsigned)(((long long)NN * H1 + T - 1) / T), T, 0, stream>>>(agg, (long long)NN * H1);
  k_scatter<<<(unsigned)(((long long)NE * (H1 / 8) + T - 1) / T), T, 0, stream>>>(bufB, ei, agg, NE, H1);
  k_scale_cvt<<<(unsigned)(((long long)NN * (H1 / 8) + T - 1) / T), T, 0, stream>>>(agg, deg, aggh, NN, H1);
  k_sage_gemm_norm<<<NN / 16, (H2 / 16) * 32, 0, stream>>>(bufB, aggh, w2l16, w2r16, b2, bufA, H1, H2);

  // ---- Layer 3: in bufA (128) -> out (2) + log_softmax
  k_zero_f32<<<(unsigned)(((long long)NN * H2 + T - 1) / T), T, 0, stream>>>(agg, (long long)NN * H2);
  k_scatter<<<(unsigned)(((long long)NE * (H2 / 8) + T - 1) / T), T, 0, stream>>>(bufA, ei, agg, NE, H2);
  k_scale_cvt<<<(unsigned)(((long long)NN * (H2 / 8) + T - 1) / T), T, 0, stream>>>(agg, deg, aggh, NN, H2);
  k_layer3<<<(NN + T - 1) / T, T, 0, stream>>>(bufA, aggh, W3l, W3r, b3, out, NN, H2);
}
